// BaseDepthNet_56659208569026
// MI455X (gfx1250) — compile-verified, matrix operand order checked by reading, not run on hardware
//
#include <hip/hip_runtime.h>

typedef __attribute__((ext_vector_type(16))) __bf16 v16bf;
typedef __attribute__((ext_vector_type(8)))  __bf16 v8bf;
typedef __attribute__((ext_vector_type(8)))  float  v8f;
typedef unsigned int u32x4 __attribute__((ext_vector_type(4)));
typedef int i32x4 __attribute__((ext_vector_type(4)));
typedef int i32x8 __attribute__((ext_vector_type(8)));

#define FH 32
#define FW 88
#define HW 2816   /* 32*88 */
#define PH 34
#define PW 90
#define PHW 3060  /* 34*90 zero-padded plane */
#define NCAM 6
#define DCH 112
#define BSTR 40   /* B-tile LDS row stride (elems): 80B, 16B-aligned, bank-spread */
#define BN_SCALE 0.99999500003749978f /* 1/sqrt(1+1e-5) */

#if __has_builtin(__builtin_amdgcn_tensor_load_to_lds)
#define HAVE_TDM 1
#endif

// ---------------------------------------------------------------------------
// TDM: DMA a 2D tile (tile_d0 contiguous bf16 x tile_d1 rows, row stride
// stride_elems) from global memory into LDS at lds_off. D# per ISA 8.3/8.4.
// ---------------------------------------------------------------------------
#ifdef HAVE_TDM
__device__ __forceinline__ void tdm_load_2d_bf16(unsigned lds_off,
                                                 const __bf16* gptr,
                                                 int tile_d0, int tile_d1,
                                                 unsigned tensor_d1,
                                                 unsigned long long stride_elems) {
  const unsigned long long ga = (unsigned long long)gptr;
  u32x4 g0;
  g0[0] = 1u;                                   // count=1 (valid), user mode
  g0[1] = lds_off;                              // lds_addr (bytes)
  g0[2] = (unsigned)(ga & 0xffffffffu);         // global_addr[31:0]
  g0[3] = (unsigned)((ga >> 32) & 0x1ffffffu) | (2u << 30);  // addr[56:32]|type=2
  const unsigned td0 = (unsigned)tile_d0;       // tensor_dim0 == tile width
  i32x8 g1;
  g1[0] = (int)(1u << 16);                      // workgroup_mask=0, data_size=1 (2B)
  g1[1] = (int)((td0 & 0xffffu) << 16);         // tensor_dim0[15:0]
  g1[2] = (int)(((td0 >> 16) & 0xffffu) | ((tensor_d1 & 0xffffu) << 16));
  g1[3] = (int)(((tensor_d1 >> 16) & 0xffffu) | (((unsigned)tile_d0 & 0xffffu) << 16));
  g1[4] = (int)((unsigned)tile_d1 & 0xffffu);   // tile_dim1; tile_dim2=0
  g1[5] = (int)(unsigned)(stride_elems & 0xffffffffull);     // dim0_stride[31:0]
  g1[6] = (int)(unsigned)((stride_elems >> 32) & 0xffffull); // dim0_stride[47:32]
  g1[7] = 0;
  const i32x4 z4 = {0, 0, 0, 0};
#if __clang_major__ >= 23
  const i32x8 z8 = {0, 0, 0, 0, 0, 0, 0, 0};
  __builtin_amdgcn_tensor_load_to_lds(g0, g1, z4, z4, z8, 0);
#else
  __builtin_amdgcn_tensor_load_to_lds(g0, g1, z4, z4, 0);
#endif
}
#endif

// ---------------------------------------------------------------------------
// Pad + convert input features to bf16: (6,512,32,88) f32 -> (6,512,34,90) bf16
// ---------------------------------------------------------------------------
__global__ __launch_bounds__(256)
void pad_img_kernel(const float* __restrict__ img, __bf16* __restrict__ Pimg) {
  const long long idx = (long long)blockIdx.x * 256 + threadIdx.x;
  const long long total = (long long)NCAM * 512 * PHW;
  if (idx >= total) return;
  const int p = (int)(idx % PHW);
  const long long plane = idx / PHW;  // cam*512+ci
  const int hp = p / PW, wp = p % PW;
  float v = 0.f;
  if (hp >= 1 && hp <= FH && wp >= 1 && wp <= FW)
    v = img[plane * HW + (hp - 1) * FW + (wp - 1)];
  Pimg[idx] = (__bf16)v;
}

// ---------------------------------------------------------------------------
// Weight transpose + bf16 convert: (CO,CIN,NTAP) f32 -> (NTAP,CO,CIN) bf16
// ---------------------------------------------------------------------------
__global__ __launch_bounds__(256)
void wtrans_kernel(const float* __restrict__ w, __bf16* __restrict__ wT,
                   int CO, int CIN, int NTAP) {
  const int idx = blockIdx.x * 256 + threadIdx.x;
  if (idx >= CO * CIN * NTAP) return;
  const int tap = idx / (CO * CIN);
  const int rem = idx % (CO * CIN);
  const int co = rem / CIN;
  const int ci = rem % CIN;
  wT[idx] = (__bf16)w[((size_t)co * CIN + ci) * NTAP + tap];
}

// ---------------------------------------------------------------------------
// SE conditioning MLP: one block per camera, 256 threads.
// ---------------------------------------------------------------------------
__global__ __launch_bounds__(256)
void se_kernel(const float* __restrict__ intrins,
               const float* __restrict__ fc1_w, const float* __restrict__ fc1_b,
               const float* __restrict__ fc2_w, const float* __restrict__ fc2_b,
               const float* __restrict__ se1_w, const float* __restrict__ se1_b,
               const float* __restrict__ se2_w, const float* __restrict__ se2_b,
               float* __restrict__ se_sig) {
  __shared__ float h[256];
  __shared__ float t1[256];
  __shared__ float t2[256];
  const int cam = blockIdx.x;
  const int i = threadIdx.x;
  const float fx = intrins[cam * 16 + 0];
  const float fy = intrins[cam * 16 + 5];
  const float pix = sqrtf(1.f / (fx * fx) + 1.f / (fy * fy)) * 1000.f;

  h[i] = fmaxf(pix * fc1_w[i] + fc1_b[i], 0.f);
  __syncthreads();
  float a = fc2_b[i];
  for (int j = 0; j < 256; ++j) a += fc2_w[i * 256 + j] * h[j];
  t1[i] = a;
  __syncthreads();
  a = se1_b[i];
  for (int j = 0; j < 256; ++j) a += se1_w[i * 256 + j] * t1[j];
  t2[i] = fmaxf(a, 0.f);
  __syncthreads();
  a = se2_b[i];
  for (int j = 0; j < 256; ++j) a += se2_w[i * 256 + j] * t2[j];
  se_sig[cam * 256 + i] = 1.f / (1.f + __expf(-a));
}

// ---------------------------------------------------------------------------
// Implicit-GEMM conv via WMMA bf16 over the padded pixel domain.
// Double-buffered pipelined mainloop:
//   - A tile (128co x 32k weights) DMA'd by the Tensor Data Mover (wave 0,
//     s_wait_tensorcnt before the barrier), one iteration ahead.
//   - B tile (128px x 32k im2col) global->regs one iteration ahead; LDS
//     scatter after the WMMA burst. One s_barrier per K-step.
// Block: 256 threads = 8 wave32 (4M x 2N); wave tile 32co x 64px = 8 WMMA/step.
// NOTE: all shared-memory buffer bases are computed at runtime from `smem`
// (no pointer arrays -> no addrspacecast static initializers).
// ---------------------------------------------------------------------------
template <int CIN, int NTAP, bool RELU, bool USE_SE, bool USE_SKIP, int CO,
          bool PADOUT>
__global__ __launch_bounds__(256)
void conv_wmma_kernel(const __bf16* __restrict__ in,
                      const __bf16* __restrict__ wgtT,
                      const float* __restrict__ gamma,
                      const float* __restrict__ beta,
                      const float* __restrict__ cbias,
                      const float* __restrict__ se_sig,
                      const __bf16* __restrict__ skip,
                      __bf16* __restrict__ outBf,
                      float* __restrict__ outF) {
  constexpr int CPI = CIN / 32;        // ci chunks per tap
  constexpr int NIT = NTAP * CPI;      // total K-steps
  constexpr int AELEM = 128 * 32;
  constexpr int BELEM = 128 * BSTR;

  __shared__ alignas(16) __bf16 smem[2 * AELEM + 2 * BELEM];

  const int cam = blockIdx.z;
  const int coBase = blockIdx.y * 128;
  const int pBase = blockIdx.x * 128;
  const int tid = threadIdx.x;
  const int wave = tid >> 5;
  const int lane = tid & 31;
  const int lh = lane >> 4;
  const int ll = lane & 15;
  const int waveM = wave & 3;
  const int waveN = wave >> 2;

  const __bf16* inC = in + (size_t)cam * CIN * PHW;
  const __bf16* wT = (const __bf16*)__builtin_assume_aligned(wgtT, 32);

  // B-staging mapping: kk fixed per thread; two 8-px groups per thread.
  const int kk = tid & 31;
  const int n0b = (tid >> 5) * 8;  // + {0, 64}
#ifndef HAVE_TDM
  const int arow = tid >> 1;            // manual A fallback mapping
  const int akk0 = (tid & 1) * 16;
  const int aco = coBase + arow;
#endif
#ifdef HAVE_TDM
  // runtime LDS byte offset of smem (low 32 bits of the generic address)
  const unsigned smem_lds = (unsigned)(unsigned long long)(&smem[0]);
#endif

  v8f acc[2][4] = {};
  __bf16 breg[2][8];

  // ---- stage helpers ----
  auto a_stage = [&](int it, int buf) {
    const int tap2 = it / CPI, c2 = it % CPI;
#ifdef HAVE_TDM
    if (wave == 0) {
      const __bf16* ag = wT + ((size_t)tap2 * CO + coBase) * CIN + c2 * 32;
      const unsigned lds_off = smem_lds + (unsigned)(buf * AELEM * 2);
      const unsigned rows = (unsigned)(CO - coBase < 128 ? CO - coBase : 128);
      tdm_load_2d_bf16(lds_off, ag, 32, 128, rows, (unsigned long long)CIN);
    }
#else
    __bf16* Ab = smem + buf * AELEM;
    const __bf16* asrc =
        wT + ((size_t)tap2 * CO + (aco < CO ? aco : 0)) * CIN + c2 * 32 + akk0;
#pragma unroll
    for (int i = 0; i < 16; ++i)
      Ab[arow * 32 + akk0 + i] = (aco < CO) ? asrc[i] : (__bf16)0.f;
#endif
  };
  auto b_gload = [&](int it) {
    const int tap2 = it / CPI, c2 = it % CPI;
    const int off2 = (NTAP == 9) ? ((tap2 / 3 - 1) * PW + (tap2 % 3 - 1)) : 0;
#pragma unroll
    for (int t = 0; t < 2; ++t) {
      const __bf16* bsrc =
          inC + (size_t)(c2 * 32 + kk) * PHW + (pBase + n0b + t * 64) + off2;
      __builtin_prefetch(bsrc + 32 * PHW, 0, 0);
#pragma unroll
      for (int i = 0; i < 8; ++i) breg[t][i] = bsrc[i];
    }
  };
  auto b_store = [&](int buf) {
    __bf16* Bb = smem + 2 * AELEM + buf * BELEM;
#pragma unroll
    for (int t = 0; t < 2; ++t) {
      const int n0 = n0b + t * 64;
#pragma unroll
      for (int i = 0; i < 8; ++i) Bb[(n0 + i) * BSTR + kk] = breg[t][i];
    }
  };

  // ---- prologue: fill buffer 0 ----
  a_stage(0, 0);
  b_gload(0);
  b_store(0);
#ifdef HAVE_TDM
  if (wave == 0) __builtin_amdgcn_s_wait_tensorcnt(0);
#endif
  __syncthreads();

  // ---- pipelined mainloop: one barrier per K-step ----
  for (int it = 0; it < NIT; ++it) {
    const int cur = it & 1, nxt = cur ^ 1;
    if (it + 1 < NIT) {
      a_stage(it + 1, nxt);  // TDM DMA overlaps the WMMA burst below
      b_gload(it + 1);       // global loads in flight during WMMA
    }

    // ---- fragments + 8 WMMA from buffer `cur` ----
    const __bf16* As = smem + cur * AELEM;
    const __bf16* Bs = smem + 2 * AELEM + cur * BELEM;
    v16bf af[2];
#pragma unroll
    for (int mi = 0; mi < 2; ++mi) {
      const int row = waveM * 32 + mi * 16 + ll;
      const v8bf lo = *reinterpret_cast<const v8bf*>(&As[row * 32 + lh * 8]);
      const v8bf hi = *reinterpret_cast<const v8bf*>(&As[row * 32 + 16 + lh * 8]);
      af[mi] = __builtin_shufflevector(lo, hi, 0, 1, 2, 3, 4, 5, 6, 7, 8, 9, 10,
                                       11, 12, 13, 14, 15);
    }
#pragma unroll
    for (int j = 0; j < 4; ++j) {
      const int brow = waveN * 64 + j * 16 + ll;
      const v8bf lo = *reinterpret_cast<const v8bf*>(&Bs[brow * BSTR + lh * 8]);
      const v8bf hi =
          *reinterpret_cast<const v8bf*>(&Bs[brow * BSTR + 16 + lh * 8]);
      const v16bf bfrag = __builtin_shufflevector(
          lo, hi, 0, 1, 2, 3, 4, 5, 6, 7, 8, 9, 10, 11, 12, 13, 14, 15);
#pragma unroll
      for (int mi = 0; mi < 2; ++mi)
        acc[mi][j] = __builtin_amdgcn_wmma_f32_16x16x32_bf16(
            false, af[mi], false, bfrag, (short)0, acc[mi][j], false, false);
    }

    if (it + 1 < NIT) b_store(nxt);  // LDS scatter after WMMA issue
#ifdef HAVE_TDM
    if (wave == 0) __builtin_amdgcn_s_wait_tensorcnt(0);
#endif
    __syncthreads();
  }

  // ---- epilogue: BN scale/bias (+conv bias), SE, skip, ReLU, masked store --
  __bf16* outBC = PADOUT ? (outBf + (size_t)cam * CO * PHW) : nullptr;
  float* outFC = PADOUT ? nullptr : (outF + (size_t)cam * CO * HW);
  const __bf16* skipC = USE_SKIP ? (skip + (size_t)cam * CO * PHW) : nullptr;
#pragma unroll
  for (int j = 0; j < 4; ++j) {
    const int p = pBase + waveN * 64 + j * 16 + ll;
    if (p >= PHW) continue;
    const int hp = p / PW, wp = p % PW;
    const bool real = (hp >= 1) && (hp <= FH) && (wp >= 1) && (wp <= FW);
#pragma unroll
    for (int mi = 0; mi < 2; ++mi) {
#pragma unroll
      for (int r = 0; r < 8; ++r) {
        const int co = coBase + waveM * 32 + mi * 16 + lh * 8 + r;
        if (co >= CO) continue;
        float v = acc[mi][j][r];
        const float sc = gamma ? gamma[co] * BN_SCALE : 1.f;
        float bi = beta ? beta[co] : 0.f;
        if (cbias) bi += cbias[co] * sc;
        v = v * sc + bi;
        if (USE_SE) v *= se_sig[cam * 256 + co];
        if (USE_SKIP) v += (float)skipC[(size_t)co * PHW + p];
        if (RELU) v = fmaxf(v, 0.f);
        if (PADOUT) {
          outBC[(size_t)co * PHW + p] = real ? (__bf16)v : (__bf16)0.f;
        } else if (real) {
          outFC[(size_t)co * HW + (hp - 1) * FW + (wp - 1)] = v;
        }
      }
    }
  }
}

// ---------------------------------------------------------------------------
// In-place softmax over 112 depth channels; one thread per (cam, pixel).
// ---------------------------------------------------------------------------
__global__ __launch_bounds__(256)
void softmax_kernel(float* __restrict__ logits) {
  const int idx = blockIdx.x * blockDim.x + threadIdx.x;
  if (idx >= NCAM * HW) return;
  const int cam = idx / HW;
  const int p = idx % HW;
  float* base = logits + (size_t)cam * DCH * HW + p;
  float mx = -1e30f;
  for (int d = 0; d < DCH; ++d) mx = fmaxf(mx, base[(size_t)d * HW]);
  float s = 0.f;
  for (int d = 0; d < DCH; ++d) {
    const float e = __expf(base[(size_t)d * HW] - mx);
    base[(size_t)d * HW] = e;
    s += e;
  }
  const float inv = 1.f / s;
  for (int d = 0; d < DCH; ++d) base[(size_t)d * HW] *= inv;
}

// ---------------------------------------------------------------------------
// Frustum projection + trilinear sample + sum over 6 cameras.
// ---------------------------------------------------------------------------
__global__ __launch_bounds__(256)
void voxel_kernel(const float* __restrict__ dp,
                  const float* __restrict__ intrins,
                  const float* __restrict__ ego2lidar,
                  const float* __restrict__ lidar2cam,
                  float* __restrict__ out) {
  const int idx = blockIdx.x * blockDim.x + threadIdx.x;
  if (idx >= 128 * 128 * 16) return;
  const int vz = idx & 15;
  const int vy = (idx >> 4) & 127;
  const int vx = idx >> 11;
  const float px = vx * 0.8f - 51.2f;
  const float py = vy * 0.8f - 51.2f;
  const float pz = vz * 0.5f - 5.0f;
  const float bin_size = 2.0f * (58.0f - 2.0f) / (112.0f * 113.0f);

  float accv = 0.f;
  for (int cam = 0; cam < NCAM; ++cam) {
    const float* L = lidar2cam + cam * 16;
    const float* E = ego2lidar + cam * 16;
    float T[16];
#pragma unroll
    for (int r = 0; r < 4; ++r)
#pragma unroll
      for (int c = 0; c < 4; ++c) {
        float s = 0.f;
#pragma unroll
        for (int k = 0; k < 4; ++k) s += L[r * 4 + k] * E[k * 4 + c];
        T[r * 4 + c] = s;
      }
    const float cx = T[0] * px + T[1] * py + T[2] * pz + T[3];
    const float cy = T[4] * px + T[5] * py + T[6] * pz + T[7];
    const float cz = T[8] * px + T[9] * py + T[10] * pz + T[11];
    const float cw = T[12] * px + T[13] * py + T[14] * pz + T[15];

    const float* K = intrins + cam * 16;
    const float uu = K[0] * cx + K[1] * cy + K[2] * cz + K[3] * cw;
    const float vv = K[4] * cx + K[5] * cy + K[6] * cz + K[7] * cw;
    const float dd = K[8] * cx + K[9] * cy + K[10] * cz + K[11] * cw;
    if (dd <= 1e-3f) continue;
    const float u = uu / dd;
    const float v = vv / dd;
    const float arg = 1.0f + 8.0f * (dd - 2.0f) / bin_size;
    if (arg <= 0.f) continue;
    const float dbin = -0.5f + 0.5f * sqrtf(arg);

    const int w0 = (int)floorf(u);
    const int h0 = (int)floorf(v);
    const int d0 = (int)floorf(dbin);
    const float fw = u - (float)w0;
    const float fh = v - (float)h0;
    const float fd = dbin - (float)d0;
    const float* vol = dp + (size_t)cam * DCH * HW;
#pragma unroll
    for (int dc = 0; dc < 2; ++dc)
#pragma unroll
      for (int hc = 0; hc < 2; ++hc)
#pragma unroll
        for (int wc = 0; wc < 2; ++wc) {
          const int di = d0 + dc, hi = h0 + hc, wi = w0 + wc;
          if (di < 0 || di >= DCH || hi < 0 || hi >= FH || wi < 0 || wi >= FW)
            continue;
          const float wgt = (dc ? fd : 1.f - fd) * (hc ? fh : 1.f - fh) *
                            (wc ? fw : 1.f - fw);
          accv += wgt * vol[((size_t)di * FH + hi) * FW + wi];
        }
  }
  out[idx] = accv;
}

// ---------------------------------------------------------------------------
extern "C" void kernel_launch(void* const* d_in, const int* in_sizes, int n_in,
                              void* d_out, int out_size, void* d_ws,
                              size_t ws_size, hipStream_t stream) {
  (void)in_sizes; (void)n_in; (void)out_size; (void)ws_size;
  const float* img       = (const float*)d_in[0];
  const float* intrins   = (const float*)d_in[1];
  const float* ego2lidar = (const float*)d_in[2];
  const float* lidar2cam = (const float*)d_in[3];
  const float* red_w  = (const float*)d_in[4];
  const float* red_cb = (const float*)d_in[5];
  const float* red_g  = (const float*)d_in[6];
  const float* red_b  = (const float*)d_in[7];
  const float* fc1_w  = (const float*)d_in[8];
  const float* fc1_b  = (const float*)d_in[9];
  const float* fc2_w  = (const float*)d_in[10];
  const float* fc2_b  = (const float*)d_in[11];
  const float* se1_w  = (const float*)d_in[12];
  const float* se1_b  = (const float*)d_in[13];
  const float* se2_w  = (const float*)d_in[14];
  const float* se2_b  = (const float*)d_in[15];
  const float* pred_w = (const float*)d_in[16];
  const float* pred_b = (const float*)d_in[17];

  // ---- workspace layout (f32 region, then bf16 region w/ guard bands) ----
  float* ws = (float*)d_ws;
  float* se_sig = ws;                   // 1536
  float* logits = ws + 1536;            // 6*112*2816 = 1,892,352
  __bf16* b16 = (__bf16*)(logits + 1892352);
  __bf16* Pimg = b16 + 128;             // 6*512*3060 = 9,400,320
  __bf16* bufA = Pimg + 9400320 + 128;  // 6*256*3060 = 4,700,160
  __bf16* bufB = bufA + 4700160 + 128;
  __bf16* wTred = bufB + 4700160 + 128; // 256*512*9 = 1,179,648
  __bf16* wTbb[6];
  for (int k = 0; k < 6; ++k) wTbb[k] = wTred + 1179648 + (size_t)k * 589824;
  __bf16* wTpred = wTred + 1179648 + 6 * 589824;  // 112*256

  // ---- prep ----
  pad_img_kernel<<<(NCAM * 512 * PHW + 255) / 256, 256, 0, stream>>>(img, Pimg);
  wtrans_kernel<<<(256 * 512 * 9 + 255) / 256, 256, 0, stream>>>(red_w, wTred,
                                                                 256, 512, 9);
  for (int i = 0; i < 3; ++i) {
    wtrans_kernel<<<(256 * 256 * 9 + 255) / 256, 256, 0, stream>>>(
        (const float*)d_in[18 + i * 6 + 0], wTbb[i * 2 + 0], 256, 256, 9);
    wtrans_kernel<<<(256 * 256 * 9 + 255) / 256, 256, 0, stream>>>(
        (const float*)d_in[18 + i * 6 + 3], wTbb[i * 2 + 1], 256, 256, 9);
  }
  wtrans_kernel<<<(112 * 256 + 255) / 256, 256, 0, stream>>>(pred_w, wTpred,
                                                             112, 256, 1);

  se_kernel<<<NCAM, 256, 0, stream>>>(intrins, fc1_w, fc1_b, fc2_w, fc2_b,
                                      se1_w, se1_b, se2_w, se2_b, se_sig);

  // ---- conv stack ----
  conv_wmma_kernel<512, 9, true, true, false, 256, true>
      <<<dim3(24, 2, NCAM), 256, 0, stream>>>(Pimg, wTred, red_g, red_b,
                                              red_cb, se_sig, nullptr, bufA,
                                              nullptr);
  for (int i = 0; i < 3; ++i) {
    const float* g1 = (const float*)d_in[18 + i * 6 + 1];
    const float* b1 = (const float*)d_in[18 + i * 6 + 2];
    const float* g2 = (const float*)d_in[18 + i * 6 + 4];
    const float* b2 = (const float*)d_in[18 + i * 6 + 5];
    conv_wmma_kernel<256, 9, true, false, false, 256, true>
        <<<dim3(24, 2, NCAM), 256, 0, stream>>>(bufA, wTbb[i * 2 + 0], g1, b1,
                                                nullptr, nullptr, nullptr,
                                                bufB, nullptr);
    conv_wmma_kernel<256, 9, true, false, true, 256, true>
        <<<dim3(24, 2, NCAM), 256, 0, stream>>>(bufB, wTbb[i * 2 + 1], g2, b2,
                                                nullptr, nullptr, bufA, bufA,
                                                nullptr);
  }
  conv_wmma_kernel<256, 1, false, false, false, DCH, false>
      <<<dim3(24, 1, NCAM), 256, 0, stream>>>(bufA, wTpred, nullptr, pred_b,
                                              nullptr, nullptr, nullptr,
                                              nullptr, logits);

  softmax_kernel<<<(NCAM * HW + 255) / 256, 256, 0, stream>>>(logits);
  voxel_kernel<<<(128 * 128 * 16 + 255) / 256, 256, 0, stream>>>(
      logits, intrins, ego2lidar, lidar2cam, (float*)d_out);
}
